// Mlp_78331613545116
// MI455X (gfx1250) — matrix operand order repaired and checked by reading, not compiled, not run on hardware
//
#include <hip/hip_runtime.h>

#define NEXPV 16
#define HDIM  1024
#define EDIM  1024
#define FDIM  2048
#define SEQ   1024
#define ALPHAC 1.702f
#define LIMITC 7.0f

// LDS pitches (dwords). XP*4 bytes must be a multiple of 16 for b128 loads.
#define XP 20
#define WP 132

typedef float v8f __attribute__((ext_vector_type(8)));
typedef __bf16 bf16x16 __attribute__((ext_vector_type(16)));

struct U8 { unsigned int d[8]; };

__device__ __forceinline__ void split_bf16(float f, unsigned int& hi, unsigned int& lo) {
    unsigned int u = __float_as_uint(f);
    hi = (u + 0x7FFFu + ((u >> 16) & 1u)) >> 16;          // RTNE high bf16
    float fh = __uint_as_float(hi << 16);
    float r = f - fh;                                      // exact residual
    unsigned int ur = __float_as_uint(r);
    lo = (ur + 0x7FFFu + ((ur >> 16) & 1u)) >> 16;        // RTNE low bf16
}

__device__ __forceinline__ v8f wmma_bf16(bf16x16 a, bf16x16 b, v8f c) {
    return __builtin_amdgcn_wmma_f32_16x16x32_bf16(false, a, false, b, (short)0, c, false, false);
}

// A fragment: 16x32 bf16, lane = (m = l&15, khalf = l>>4); two b128 chunks per lane.
__device__ __forceinline__ bf16x16 load_afrag(const unsigned int* lds, int row, int hl) {
    U8 u;
    const uint4 q0 = *(const uint4*)(lds + row * XP + hl * 8);
    const uint4 q1 = *(const uint4*)(lds + row * XP + hl * 8 + 4);
    u.d[0] = q0.x; u.d[1] = q0.y; u.d[2] = q0.z; u.d[3] = q0.w;
    u.d[4] = q1.x; u.d[5] = q1.y; u.d[6] = q1.z; u.d[7] = q1.w;
    return __builtin_bit_cast(bf16x16, u);
}

// B fragment: 32x16 bf16, lane = (n = l&15, khalf = l>>4); LDS holds k-pair-packed dwords [k/2][n].
__device__ __forceinline__ bf16x16 load_bfrag(const unsigned int* lds, int col, int hl) {
    U8 u;
    const unsigned int* p = lds + hl * 8 * WP + col;
#pragma unroll
    for (int j = 0; j < 8; ++j) u.d[j] = p[j * WP];
    return __builtin_bit_cast(bf16x16, u);
}

// ---------------- Phase 0: zero counts ----------------
__global__ void zero_kernel(int* counts) {
    if (threadIdx.x < NEXPV) counts[threadIdx.x] = 0;
}

// ---------------- Phase 1: router ----------------
__global__ __launch_bounds__(256) void router_kernel(
    const float* __restrict__ x, const float* __restrict__ Wg, const float* __restrict__ bg,
    int* __restrict__ counts, int* __restrict__ topidx, float* __restrict__ topw) {
    __shared__ float red[256][17];
    const int t = blockIdx.x, tid = threadIdx.x;
    const float4 xv = *(const float4*)(x + (size_t)t * HDIM + tid * 4);
#pragma unroll
    for (int e = 0; e < NEXPV; ++e) {
        const float* wr = Wg + e * HDIM + tid * 4;
        red[tid][e] = xv.x * wr[0] + xv.y * wr[1] + xv.z * wr[2] + xv.w * wr[3];
    }
    __syncthreads();
    for (int s = 128; s > 0; s >>= 1) {
        if (tid < s) {
#pragma unroll
            for (int e = 0; e < NEXPV; ++e) red[tid][e] += red[tid + s][e];
        }
        __syncthreads();
    }
    if (tid == 0) {
        float lg[NEXPV];
#pragma unroll
        for (int e = 0; e < NEXPV; ++e) lg[e] = red[0][e] + bg[e];
        int i0 = 0;
        for (int e = 1; e < NEXPV; ++e) if (lg[e] > lg[i0]) i0 = e;
        int i1 = (i0 == 0) ? 1 : 0;
        for (int e = 0; e < NEXPV; ++e) if (e != i0 && lg[e] > lg[i1]) i1 = e;
        float d = __expf(lg[i1] - lg[i0]);
        float w0 = 1.0f / (1.0f + d);
        float w1 = 1.0f - w0;
        topidx[2 * t] = i0; topidx[2 * t + 1] = i1;
        topw[2 * t] = w0;  topw[2 * t + 1] = w1;
        atomicAdd(&counts[i0], 1);
        atomicAdd(&counts[i1], 1);
    }
}

// ---------------- Phase 2: scan ----------------
__global__ void scan_kernel(const int* __restrict__ counts, int* __restrict__ base, int* __restrict__ cursor) {
    if (threadIdx.x == 0) {
        int s = 0;
        for (int e = 0; e < NEXPV; ++e) { base[e] = s; cursor[e] = s; s += counts[e]; }
        base[NEXPV] = s;
    }
}

// ---------------- Phase 3: slot assignment ----------------
__global__ void assign_kernel(const int* __restrict__ topidx, const float* __restrict__ topw,
                              int* __restrict__ cursor, int* __restrict__ slot_token,
                              float* __restrict__ slot_w, int* __restrict__ token_slot) {
    int t = blockIdx.x * 256 + threadIdx.x;
    if (t >= SEQ) return;
#pragma unroll
    for (int k = 0; k < 2; ++k) {
        int e = topidx[2 * t + k];
        float w = topw[2 * t + k];
        int slot = atomicAdd(&cursor[e], 1);
        slot_token[slot] = t;
        slot_w[slot] = w;
        token_slot[2 * t + k] = slot;
    }
}

// ---------------- Phase 4: gate/up GEMM + activation ----------------
// grid: (f-tile 0..15, m-tile 0..15, expert 0..15); block 256 = 8 waves (2 m x 4 n).
__global__ __launch_bounds__(256) void gemm1_kernel(
    const float* __restrict__ x, const float* __restrict__ Wgu, const float* __restrict__ bgu,
    const int* __restrict__ counts, const int* __restrict__ base,
    const int* __restrict__ slot_token, const float* __restrict__ slot_w,
    float* __restrict__ gated_ws) {
    const int e = blockIdx.z;
    const int cnt = counts[e];
    const int m0 = blockIdx.y * 64;
    if (m0 >= cnt) return;
    const int n0 = blockIdx.x * 128;
    const int sbase = base[e];

    __shared__ __align__(16) unsigned int lds_xh[64 * XP], lds_xl[64 * XP];
    __shared__ __align__(16) unsigned int lds_wh[16 * WP], lds_wl[16 * WP];
    __shared__ int   lds_tok[64];
    __shared__ float lds_wt[64];

    const int tid = threadIdx.x;
    if (tid < 64) {
        int mi = m0 + tid;
        bool valid = mi < cnt;
        int slot = sbase + (valid ? mi : 0);
        lds_tok[tid] = slot_token[slot];
        lds_wt[tid]  = valid ? slot_w[slot] : 0.0f;
    }
    __syncthreads();

    const int wid = tid >> 5, lane = tid & 31;
    const int wm = wid >> 2, wn = wid & 3, ml = lane & 15, hl = lane >> 4;
    const float* wbase = Wgu + (size_t)e * HDIM * FDIM + n0;

    v8f acc[2][2];
#pragma unroll
    for (int i = 0; i < 2; ++i)
#pragma unroll
        for (int j = 0; j < 2; ++j) acc[i][j] = (v8f){0,0,0,0,0,0,0,0};

    for (int k0 = 0; k0 < HDIM; k0 += 32) {
        // Stage X tile (64 rows x 32 k), split to bf16 hi/lo, k-pairs packed in dwords.
#pragma unroll
        for (int i = 0; i < 2; ++i) {
            int tt = tid + 256 * i;
            int r = tt >> 3, c = tt & 7;
            const float4 xv = *(const float4*)(x + (size_t)lds_tok[r] * HDIM + k0 + c * 4);
            unsigned int h0,h1,h2,h3,l0,l1,l2,l3;
            split_bf16(xv.x,h0,l0); split_bf16(xv.y,h1,l1);
            split_bf16(xv.z,h2,l2); split_bf16(xv.w,h3,l3);
            lds_xh[r * XP + c * 2]     = (h1 << 16) | h0;
            lds_xh[r * XP + c * 2 + 1] = (h3 << 16) | h2;
            lds_xl[r * XP + c * 2]     = (l1 << 16) | l0;
            lds_xl[r * XP + c * 2 + 1] = (l3 << 16) | l2;
        }
        // Stage W tile (32 k x 128 n) as k-pair-packed dwords [k/2][n].
#pragma unroll
        for (int i = 0; i < 2; ++i) {
            int tt = tid + 256 * i;
            int p = tt >> 5, c = tt & 31;
            const float* ra = wbase + (size_t)(k0 + 2 * p) * FDIM + c * 4;
            const float4 av = *(const float4*)ra;
            const float4 bv = *(const float4*)(ra + FDIM);
            unsigned int ah[4], al[4], bh[4], bl[4];
            split_bf16(av.x,ah[0],al[0]); split_bf16(av.y,ah[1],al[1]);
            split_bf16(av.z,ah[2],al[2]); split_bf16(av.w,ah[3],al[3]);
            split_bf16(bv.x,bh[0],bl[0]); split_bf16(bv.y,bh[1],bl[1]);
            split_bf16(bv.z,bh[2],bl[2]); split_bf16(bv.w,bh[3],bl[3]);
            uint4 ph, pl;
            ph.x = (bh[0]<<16)|ah[0]; ph.y = (bh[1]<<16)|ah[1];
            ph.z = (bh[2]<<16)|ah[2]; ph.w = (bh[3]<<16)|ah[3];
            pl.x = (bl[0]<<16)|al[0]; pl.y = (bl[1]<<16)|al[1];
            pl.z = (bl[2]<<16)|al[2]; pl.w = (bl[3]<<16)|al[3];
            *(uint4*)&lds_wh[p * WP + c * 4] = ph;
            *(uint4*)&lds_wl[p * WP + c * 4] = pl;
        }
        __syncthreads();

        const int r0 = wm * 32 + ml, r1 = r0 + 16;
        bf16x16 ah0 = load_afrag(lds_xh, r0, hl), ah1 = load_afrag(lds_xh, r1, hl);
        bf16x16 al0 = load_afrag(lds_xl, r0, hl), al1 = load_afrag(lds_xl, r1, hl);
        const int c0 = wn * 32 + ml, c1 = c0 + 16;
        bf16x16 bh0 = load_bfrag(lds_wh, c0, hl), bh1 = load_bfrag(lds_wh, c1, hl);
        bf16x16 bl0 = load_bfrag(lds_wl, c0, hl), bl1 = load_bfrag(lds_wl, c1, hl);

        acc[0][0] = wmma_bf16(ah0, bh0, acc[0][0]);
        acc[0][0] = wmma_bf16(al0, bh0, acc[0][0]);
        acc[0][0] = wmma_bf16(ah0, bl0, acc[0][0]);
        acc[0][1] = wmma_bf16(ah0, bh1, acc[0][1]);
        acc[0][1] = wmma_bf16(al0, bh1, acc[0][1]);
        acc[0][1] = wmma_bf16(ah0, bl1, acc[0][1]);
        acc[1][0] = wmma_bf16(ah1, bh0, acc[1][0]);
        acc[1][0] = wmma_bf16(al1, bh0, acc[1][0]);
        acc[1][0] = wmma_bf16(ah1, bl0, acc[1][0]);
        acc[1][1] = wmma_bf16(ah1, bh1, acc[1][1]);
        acc[1][1] = wmma_bf16(al1, bh1, acc[1][1]);
        acc[1][1] = wmma_bf16(ah1, bl1, acc[1][1]);
        __syncthreads();
    }

    // Epilogue: bias, gate/up pairing via lane shuffle, clipped-SiLU, routing-weight scale.
#pragma unroll
    for (int ns = 0; ns < 2; ++ns) {
        const int f = n0 + wn * 32 + ns * 16 + ml;
        const float bias = bgu[e * FDIM + f];
        const bool isgate = (f & 1) == 0;
#pragma unroll
        for (int ms = 0; ms < 2; ++ms) {
#pragma unroll
            for (int v = 0; v < 8; ++v) {
                float val = acc[ms][ns][v] + bias;
                float other = __shfl_xor(val, 1, 32);
                float gate = isgate ? val : other;
                float up   = isgate ? other : val;
                gate = fminf(gate, LIMITC);
                up   = fminf(fmaxf(up, -LIMITC), LIMITC);
                float glu = gate / (1.0f + __expf(-ALPHAC * gate));
                float gated = (up + 1.0f) * glu;
                int m = wm * 32 + ms * 16 + hl * 8 + v;
                if (isgate && (m0 + m) < cnt)
                    gated_ws[(size_t)(sbase + m0 + m) * EDIM + (f >> 1)] = gated * lds_wt[m];
            }
        }
    }
}

// ---------------- Phase 5: down GEMM ----------------
// grid: (h-tile 0..7, m-tile 0..15, expert 0..15)
__global__ __launch_bounds__(256) void gemm2_kernel(
    const float* __restrict__ gated_ws, const float* __restrict__ Wd, const float* __restrict__ bd,
    const int* __restrict__ counts, const int* __restrict__ base,
    const float* __restrict__ slot_w, float* __restrict__ oute) {
    const int e = blockIdx.z;
    const int cnt = counts[e];
    const int m0 = blockIdx.y * 64;
    if (m0 >= cnt) return;
    const int n0 = blockIdx.x * 128;
    const int sbase = base[e];

    __shared__ __align__(16) unsigned int lds_xh[64 * XP], lds_xl[64 * XP];
    __shared__ __align__(16) unsigned int lds_wh[16 * WP], lds_wl[16 * WP];
    __shared__ float lds_wt[64];

    const int tid = threadIdx.x;
    if (tid < 64) {
        int mi = m0 + tid;
        bool valid = mi < cnt;
        lds_wt[tid] = valid ? slot_w[sbase + mi] : 0.0f;
    }
    __syncthreads();

    const int wid = tid >> 5, lane = tid & 31;
    const int wm = wid >> 2, wn = wid & 3, ml = lane & 15, hl = lane >> 4;
    const float* wbase = Wd + (size_t)e * EDIM * HDIM + n0;

    v8f acc[2][2];
#pragma unroll
    for (int i = 0; i < 2; ++i)
#pragma unroll
        for (int j = 0; j < 2; ++j) acc[i][j] = (v8f){0,0,0,0,0,0,0,0};

    for (int k0 = 0; k0 < EDIM; k0 += 32) {
#pragma unroll
        for (int i = 0; i < 2; ++i) {
            int tt = tid + 256 * i;
            int r = tt >> 3, c = tt & 7;
            int rr = m0 + r; if (rr >= cnt) rr = cnt - 1;   // clamp pad rows to valid memory
            const float4 xv = *(const float4*)(gated_ws + (size_t)(sbase + rr) * EDIM + k0 + c * 4);
            unsigned int h0,h1,h2,h3,l0,l1,l2,l3;
            split_bf16(xv.x,h0,l0); split_bf16(xv.y,h1,l1);
            split_bf16(xv.z,h2,l2); split_bf16(xv.w,h3,l3);
            lds_xh[r * XP + c * 2]     = (h1 << 16) | h0;
            lds_xh[r * XP + c * 2 + 1] = (h3 << 16) | h2;
            lds_xl[r * XP + c * 2]     = (l1 << 16) | l0;
            lds_xl[r * XP + c * 2 + 1] = (l3 << 16) | l2;
        }
#pragma unroll
        for (int i = 0; i < 2; ++i) {
            int tt = tid + 256 * i;
            int p = tt >> 5, c = tt & 31;
            const float* ra = wbase + (size_t)(k0 + 2 * p) * HDIM + c * 4;
            const float4 av = *(const float4*)ra;
            const float4 bv = *(const float4*)(ra + HDIM);
            unsigned int ah[4], al[4], bh[4], bl[4];
            split_bf16(av.x,ah[0],al[0]); split_bf16(av.y,ah[1],al[1]);
            split_bf16(av.z,ah[2],al[2]); split_bf16(av.w,ah[3],al[3]);
            split_bf16(bv.x,bh[0],bl[0]); split_bf16(bv.y,bh[1],bl[1]);
            split_bf16(bv.z,bh[2],bl[2]); split_bf16(bv.w,bh[3],bl[3]);
            uint4 ph, pl;
            ph.x = (bh[0]<<16)|ah[0]; ph.y = (bh[1]<<16)|ah[1];
            ph.z = (bh[2]<<16)|ah[2]; ph.w = (bh[3]<<16)|ah[3];
            pl.x = (bl[0]<<16)|al[0]; pl.y = (bl[1]<<16)|al[1];
            pl.z = (bl[2]<<16)|al[2]; pl.w = (bl[3]<<16)|al[3];
            *(uint4*)&lds_wh[p * WP + c * 4] = ph;
            *(uint4*)&lds_wl[p * WP + c * 4] = pl;
        }
        __syncthreads();

        const int r0 = wm * 32 + ml, r1 = r0 + 16;
        bf16x16 ah0 = load_afrag(lds_xh, r0, hl), ah1 = load_afrag(lds_xh, r1, hl);
        bf16x16 al0 = load_afrag(lds_xl, r0, hl), al1 = load_afrag(lds_xl, r1, hl);
        const int c0 = wn * 32 + ml, c1 = c0 + 16;
        bf16x16 bh0 = load_bfrag(lds_wh, c0, hl), bh1 = load_bfrag(lds_wh, c1, hl);
        bf16x16 bl0 = load_bfrag(lds_wl, c0, hl), bl1 = load_bfrag(lds_wl, c1, hl);

        acc[0][0] = wmma_bf16(ah0, bh0, acc[0][0]);
        acc[0][0] = wmma_bf16(al0, bh0, acc[0][0]);
        acc[0][0] = wmma_bf16(ah0, bl0, acc[0][0]);
        acc[0][1] = wmma_bf16(ah0, bh1, acc[0][1]);
        acc[0][1] = wmma_bf16(al0, bh1, acc[0][1]);
        acc[0][1] = wmma_bf16(ah0, bl1, acc[0][1]);
        acc[1][0] = wmma_bf16(ah1, bh0, acc[1][0]);
        acc[1][0] = wmma_bf16(al1, bh0, acc[1][0]);
        acc[1][0] = wmma_bf16(ah1, bl0, acc[1][0]);
        acc[1][1] = wmma_bf16(ah1, bh1, acc[1][1]);
        acc[1][1] = wmma_bf16(al1, bh1, acc[1][1]);
        acc[1][1] = wmma_bf16(ah1, bl1, acc[1][1]);
        __syncthreads();
    }

    // Epilogue: add routing-weighted bias, store per-slot expert output row.
#pragma unroll
    for (int ns = 0; ns < 2; ++ns) {
        const int h = n0 + wn * 32 + ns * 16 + ml;
        const float bias = bd[e * HDIM + h];
#pragma unroll
        for (int ms = 0; ms < 2; ++ms) {
#pragma unroll
            for (int v = 0; v < 8; ++v) {
                int m = wm * 32 + ms * 16 + hl * 8 + v;
                if ((m0 + m) < cnt)
                    oute[(size_t)(sbase + m0 + m) * HDIM + h] = acc[ms][ns][v] + lds_wt[m] * bias;
            }
        }
    }
}

// ---------------- Phase 6: combine top-2 expert outputs ----------------
__global__ void gather_kernel(const float* __restrict__ oute, const int* __restrict__ token_slot,
                              float* __restrict__ out) {
    int idx = blockIdx.x * 256 + threadIdx.x;
    if (idx >= SEQ * HDIM) return;
    int t = idx >> 10, h = idx & (HDIM - 1);
    int s0 = token_slot[2 * t], s1 = token_slot[2 * t + 1];
    out[idx] = oute[(size_t)s0 * HDIM + h] + oute[(size_t)s1 * HDIM + h];
}

extern "C" void kernel_launch(void* const* d_in, const int* in_sizes, int n_in,
                              void* d_out, int out_size, void* d_ws, size_t ws_size,
                              hipStream_t stream) {
    const float* x   = (const float*)d_in[0];
    const float* Wg  = (const float*)d_in[1];
    const float* bg  = (const float*)d_in[2];
    const float* Wgu = (const float*)d_in[3];
    const float* bgu = (const float*)d_in[4];
    const float* Wd  = (const float*)d_in[5];
    const float* bd  = (const float*)d_in[6];
    float* out = (float*)d_out;

    unsigned int* ws = (unsigned int*)d_ws;   // dword-addressed workspace (~16.8 MB used)
    int*   counts     = (int*)(ws + 0);            // 16
    int*   base       = (int*)(ws + 16);           // 17
    int*   cursor     = (int*)(ws + 40);           // 16
    int*   topidx     = (int*)(ws + 64);           // 2048
    float* topw       = (float*)(ws + 64 + 2048);  // 2048
    int*   slot_token = (int*)(ws + 64 + 4096);    // 2048
    float* slot_w     = (float*)(ws + 64 + 6144);  // 2048
    int*   token_slot = (int*)(ws + 64 + 8192);    // 2048
    float* gated      = (float*)(ws + 16384);                      // 2048*1024 f32
    float* oute       = (float*)(ws + 16384 + 2048 * 1024);        // 2048*1024 f32

    zero_kernel<<<1, 32, 0, stream>>>(counts);
    router_kernel<<<SEQ, 256, 0, stream>>>(x, Wg, bg, counts, topidx, topw);
    scan_kernel<<<1, 32, 0, stream>>>(counts, base, cursor);
    assign_kernel<<<4, 256, 0, stream>>>(topidx, topw, cursor, slot_token, slot_w, token_slot);
    gemm1_kernel<<<dim3(16, 16, 16), 256, 0, stream>>>(x, Wgu, bgu, counts, base,
                                                       slot_token, slot_w, gated);
    gemm2_kernel<<<dim3(8, 16, 16), 256, 0, stream>>>(gated, Wd, bd, counts, base, slot_w, oute);
    gather_kernel<<<(SEQ * HDIM) / 256, 256, 0, stream>>>(oute, token_slot, out);
}
